// TransMIL_28157805592807
// MI455X (gfx1250) — matrix operand order repaired and checked by reading, not compiled, not run on hardware
//
#include <hip/hip_runtime.h>
#include <hip/hip_bf16.h>

typedef __attribute__((ext_vector_type(16))) _Float16 v16h;
typedef __attribute__((ext_vector_type(8)))  _Float16 v8h;
typedef __attribute__((ext_vector_type(8)))  float    v8f;
typedef __attribute__((ext_vector_type(4)))  int      i32x4;

#define NB    4
#define TSEQ  4095
#define INDIM 1024
#define DIMC  512
#define NSEQ  4096
#define NH    8
#define DHD   64
#define NLM   256
#define LSEG  16
#define KSZ   33

#ifndef __has_builtin
#define __has_builtin(x) 0
#endif

// gfx1250 async global->LDS path (ASYNCcnt-tracked), with safe fallback
#if defined(__HIP_DEVICE_COMPILE__) && __has_builtin(__builtin_amdgcn_global_load_async_to_lds_b128)
#define ASYNC_LDS 1
#else
#define ASYNC_LDS 0
#endif

#if ASYNC_LDS
#if __has_builtin(__builtin_amdgcn_s_wait_asynccnt)
#define WAIT_ASYNC(n) __builtin_amdgcn_s_wait_asynccnt(n)
#else
#define WAIT_ASYNC(n) asm volatile("s_wait_asynccnt %0" ::"n"(n) : "memory")
#endif
#endif

__device__ __forceinline__ void copy16_g2l(const _Float16* __restrict__ g,
                                           _Float16* __restrict__ l)
{
#if ASYNC_LDS
    // builtin signature: (int4 AS1*, int4 AS3*, imm offset, imm cpol)
    __builtin_amdgcn_global_load_async_to_lds_b128(
        (__attribute__((address_space(1))) i32x4*)(_Float16*)g,
        (__attribute__((address_space(3))) i32x4*)l, 0, 0);
#else
    __builtin_prefetch(g + 256, 0, 1);          // global_prefetch_b8
    *(v8h*)l = *(const v8h*)g;
#endif
}

// ---------------------------------------------------------------------------
// Batched WMMA GEMM:  C[M,N] = alpha * A[M,K] @ B
//   BT=true : B given as Bt[N,K] row-major (weightT / k-matrix layouts)
//   BT=false: B given as B[K,N] row-major (transposed while staging to LDS)
//   OUT16   : store C as f16, else f32
// Block: 256 threads = 8 waves -> 128x64 block tile; wave tile 16x64
// (4 WMMA per K-step, A-fragment reused 4x). K staged in 32-wide panels
// through double-buffered LDS; panels loaded with async global->LDS when
// available (next panel prefetched while current is consumed by WMMA).
// Requires M%128==0, N%64==0, K%32==0, K>=64 (all shapes here comply).
// ---------------------------------------------------------------------------
template<bool BT, bool OUT16>
__global__ __launch_bounds__(256)
void wmma_gemm(const _Float16* __restrict__ A, const _Float16* __restrict__ B,
               void* __restrict__ C, int M, int N, int K,
               int lda, int ldb, int ldc,
               long long strA, long long strB, long long strC, float alpha)
{
    __shared__ _Float16 shA[2][128 * 32];   // 8 KB per buffer
    __shared__ _Float16 shB[2][64 * 32];    // 4 KB per buffer

    const int batch = blockIdx.z;
    const _Float16* Ab = A + (long long)batch * strA;
    const _Float16* Bb = B + (long long)batch * strB;

    const int tid    = threadIdx.x;
    const int wave   = tid >> 5;
    const int lane   = tid & 31;
    const int rowblk = blockIdx.y * 128;
    const int colblk = blockIdx.x * 64;
    const int r      = lane & 15;            // row (A) / col (B) within tile
    const int hi     = lane >> 4;            // K-half select
    (void)M; (void)N;

    v8f acc[4];
#pragma unroll
    for (int t = 0; t < 4; ++t)
#pragma unroll
        for (int j = 0; j < 8; ++j) acc[t][j] = 0.0f;

    // Cooperative panel staging: A 128x32 (512 x 16B chunks, 2/thread),
    // BT B-panel 64x32 (256 chunks, 1/thread), BN B-panel transposed in
    // flight with scalar loads (8/thread).
    auto stage = [&](int buf, int k) {
#pragma unroll
        for (int s = 0; s < 2; ++s) {
            const int cc = tid + s * 256;
            const int row = cc >> 2, q = cc & 3;
            copy16_g2l(Ab + (long long)(rowblk + row) * lda + k + q * 8,
                       &shA[buf][row * 32 + q * 8]);
        }
        if (BT) {
            const int col = tid >> 2, q = tid & 3;
            copy16_g2l(Bb + (long long)(colblk + col) * ldb + k + q * 8,
                       &shB[buf][col * 32 + q * 8]);
        } else {
#pragma unroll
            for (int s = 0; s < 8; ++s) {
                const int cc = tid + s * 256;
                const int kk = cc >> 6, col = cc & 63;
                shB[buf][col * 32 + kk] =
                    Bb[(long long)(k + kk) * ldb + colblk + col];
            }
        }
    };

    stage(0, 0);
    int cur = 0;
    for (int k = 0; k < K; k += 32) {
        const bool has_next = (k + 32) < K;
        if (has_next) stage(cur ^ 1, k + 32);   // prefetch next panel
#if ASYNC_LDS
        // wait for the *current* panel's async copies (in-order per wave)
        if (has_next) { WAIT_ASYNC(BT ? 3 : 2); } else { WAIT_ASYNC(0); }
#endif
        __syncthreads();

        const _Float16* As = shA[cur];
        const _Float16* Bs = shB[cur];
        // A fragment: lanes 0-15 -> K{0-7,16-23}, lanes 16-31 -> K{8-15,24-31}
        union { v16h v; v8h h[2]; } af;
        af.h[0] = *(const v8h*)(As + (wave * 16 + r) * 32 + 8 * hi);
        af.h[1] = *(const v8h*)(As + (wave * 16 + r) * 32 + 16 + 8 * hi);
#pragma unroll
        for (int t = 0; t < 4; ++t) {
            union { v16h v; v8h h[2]; } bf;
            const _Float16* bp = Bs + (t * 16 + r) * 32 + 8 * hi;
            bf.h[0] = *(const v8h*)(bp);
            bf.h[1] = *(const v8h*)(bp + 16);
            acc[t] = __builtin_amdgcn_wmma_f32_16x16x32_f16(
                false, af.v, false, bf.v, (short)0, acc[t], false, false);
        }
        __syncthreads();          // all reads done before panel is overwritten
        cur ^= 1;
    }

    // C/D layout: lane%16 = col, VGPR j -> row j (lanes 0-15) / j+8 (16-31)
#pragma unroll
    for (int t = 0; t < 4; ++t) {
        const int col = colblk + t * 16 + r;
#pragma unroll
        for (int j = 0; j < 8; ++j) {
            const int row = rowblk + wave * 16 + j + 8 * hi;
            const float val = acc[t][j] * alpha;
            const long long idx = (long long)batch * strC + (long long)row * ldc + col;
            if (OUT16) ((_Float16*)C)[idx] = (_Float16)val;
            else       ((float*)C)[idx]    = val;
        }
    }
}

// ---------------------------------------------------------------------------
// Pointwise / reduction kernels
// ---------------------------------------------------------------------------

// data [B,4095,1024] f32 -> Xin [B,4096,1024] f16, padded row 0 = zeros (cls slot)
__global__ void pack_input(const float* __restrict__ data, _Float16* __restrict__ Xin)
{
    long long i = (long long)blockIdx.x * 256 + threadIdx.x;   // B*4096*1024
    int c = (int)(i & 1023);
    long long row = i >> 10;
    int p = (int)(row & 4095);
    int b = (int)(row >> 12);
    _Float16 v = (_Float16)0.0f;
    if (p != 0) v = (_Float16)data[((long long)b * TSEQ + (p - 1)) * INDIM + c];
    Xin[i] = v;
}

// W[K,N] f32 -> WT[N,K] f16
__global__ void convert_transpose(const float* __restrict__ W, _Float16* __restrict__ WT,
                                  int K, int N)
{
    long long i = (long long)blockIdx.x * 256 + threadIdx.x;   // K*N
    int k = (int)(i / N), n = (int)(i % N);
    WT[(long long)n * K + k] = (_Float16)W[i];
}

// fc1 epilogue: row 0 = cls token; else relu(C + b) + sinusoidal PE(p-1)
__global__ void fc1_post(const float* __restrict__ C, const float* __restrict__ b1,
                         const float* __restrict__ cls, float* __restrict__ h)
{
    long long i = (long long)blockIdx.x * 256 + threadIdx.x;   // B*4096*512
    int c = (int)(i & 511);
    long long row = i >> 9;
    int p = (int)(row & 4095);
    float v;
    if (p == 0) {
        v = cls[c];
    } else {
        v = C[i] + b1[c];
        v = v > 0.0f ? v : 0.0f;
        float t = (float)(p - 1);
        float div = __expf(-__logf(10000.0f) * (float)(2 * (c >> 1)) / (float)DIMC);
        float ang = t * div;
        v += (c & 1) ? __cosf(ang) : __sinf(ang);
    }
    h[i] = v;
}

// wave-per-row layernorm: X f32 [rows,512] -> Y f16
__global__ __launch_bounds__(256)
void layernorm_f16(const float* __restrict__ X, const float* __restrict__ sc,
                   const float* __restrict__ bi, _Float16* __restrict__ Y)
{
    int wave = threadIdx.x >> 5, lane = threadIdx.x & 31;
    long long row = (long long)blockIdx.x * 8 + wave;
    const float* x = X + row * DIMC;
    float s = 0.0f, s2 = 0.0f, vals[16];
#pragma unroll
    for (int j = 0; j < 16; ++j) {
        float v = x[lane * 16 + j];
        vals[j] = v; s += v; s2 += v * v;
    }
    for (int o = 16; o > 0; o >>= 1) { s += __shfl_xor(s, o, 32); s2 += __shfl_xor(s2, o, 32); }
    float mu = s / (float)DIMC;
    float var = s2 / (float)DIMC - mu * mu;
    float rs = rsqrtf(var + 1e-5f);
    _Float16* y = Y + row * DIMC;
#pragma unroll
    for (int j = 0; j < 16; ++j) {
        int c = lane * 16 + j;
        y[c] = (_Float16)((vals[j] - mu) * rs * sc[c] + bi[c]);
    }
}

// qkv [B,4096,1536] f16 -> q,k,v [B,H,4096,64] f16 (q scaled by dh^-0.5)
__global__ void split_qkv(const _Float16* __restrict__ qkv, _Float16* __restrict__ q,
                          _Float16* __restrict__ k, _Float16* __restrict__ v)
{
    long long i = (long long)blockIdx.x * 256 + threadIdx.x;   // B*H*4096*64
    int d  = (int)(i & 63);
    int n  = (int)((i >> 6) & 4095);
    int hh = (int)((i >> 18) & 7);
    int b  = (int)(i >> 21);
    long long base = ((long long)b * NSEQ + n) * (3 * DIMC) + hh * DHD + d;
    q[i] = (_Float16)((float)qkv[base] * 0.125f);
    k[i] = qkv[base + DIMC];
    v[i] = qkv[base + 2 * DIMC];
}

// landmark means over segments of 16 tokens
__global__ void landmarks(const _Float16* __restrict__ q, const _Float16* __restrict__ k,
                          _Float16* __restrict__ ql, _Float16* __restrict__ kl)
{
    long long i = (long long)blockIdx.x * 256 + threadIdx.x;   // 32*256*64
    int d  = (int)(i & 63);
    int m  = (int)((i >> 6) & 255);
    int bh = (int)(i >> 14);
    float sq = 0.0f, sk = 0.0f;
    long long base = ((long long)bh * NSEQ + m * LSEG) * DHD + d;
#pragma unroll
    for (int j = 0; j < LSEG; ++j) {
        sq += (float)q[base + (long long)j * DHD];
        sk += (float)k[base + (long long)j * DHD];
    }
    ql[i] = (_Float16)(sq * (1.0f / LSEG));
    kl[i] = (_Float16)(sk * (1.0f / LSEG));
}

// in-place row softmax on f16, one block per row
__global__ __launch_bounds__(256)
void softmax_rows(_Float16* __restrict__ X, int rowlen)
{
    __shared__ float red[256];
    _Float16* p = X + (long long)blockIdx.x * rowlen;
    int t = threadIdx.x;
    float m = -1e30f;
    for (int i = t; i < rowlen; i += 256) m = fmaxf(m, (float)p[i]);
    red[t] = m; __syncthreads();
    for (int s = 128; s > 0; s >>= 1) { if (t < s) red[t] = fmaxf(red[t], red[t + s]); __syncthreads(); }
    m = red[0]; __syncthreads();
    float sum = 0.0f;
    for (int i = t; i < rowlen; i += 256) {
        float e = __expf((float)p[i] - m);
        p[i] = (_Float16)e; sum += e;
    }
    red[t] = sum; __syncthreads();
    for (int s = 128; s > 0; s >>= 1) { if (t < s) red[t] += red[t + s]; __syncthreads(); }
    float inv = 1.0f / red[0];
    for (int i = t; i < rowlen; i += 256) p[i] = (_Float16)((float)p[i] * inv);
}

__global__ void init_denom(float* d) { ((unsigned int*)d)[0] = 0u; }

// global max column-sum of attn2 (row sums are 1 after softmax)
__global__ void colsum_max(const _Float16* __restrict__ x2, float* __restrict__ denom)
{
    int i = blockIdx.x * 256 + threadIdx.x;                    // 32*256 threads
    int col = i & 255, bh = i >> 8;
    const _Float16* p = x2 + (long long)bh * NLM * NLM + col;
    float s = 0.0f;
    for (int r = 0; r < NLM; ++r) s += (float)p[(long long)r * NLM];
    atomicMax((unsigned int*)denom, __float_as_uint(s));       // positive floats
}

// z0 = x^T / denom
__global__ void pinv_init(const _Float16* __restrict__ x2, const float* __restrict__ denom,
                          _Float16* __restrict__ z)
{
    long long i = (long long)blockIdx.x * 256 + threadIdx.x;   // 32*256*256
    int j = (int)(i & 255), r = (int)((i >> 8) & 255), bh = (int)(i >> 16);
    float d = __uint_as_float(((const unsigned int*)denom)[0]);
    float inv = 1.0f / fmaxf(d, 1e-20f);
    z[i] = (_Float16)((float)x2[((long long)bh * NLM + j) * NLM + r] * inv);
}

// out = c*I - in (per 256x256 matrix, batched)
__global__ void diag_sub(const _Float16* __restrict__ in, _Float16* __restrict__ out, float c)
{
    long long i = (long long)blockIdx.x * 256 + threadIdx.x;   // 32*256*256
    int j = (int)(i & 255), r = (int)((i >> 8) & 255);
    out[i] = (_Float16)(((r == j) ? c : 0.0f) - (float)in[i]);
}

// depthwise conv (kernel 33, pad 16) over sequence of v, accumulated into out
__global__ void conv_add(const _Float16* __restrict__ v, const float* __restrict__ w,
                         float* __restrict__ out)
{
    long long i = (long long)blockIdx.x * 256 + threadIdx.x;   // 32*4096*64
    int d  = (int)(i & 63);
    int n  = (int)((i >> 6) & 4095);
    int bh = (int)(i >> 18);
    int hh = bh & 7;
    float acc = out[i];
    const _Float16* vb = v + ((long long)bh * NSEQ) * DHD + d;
    const float* wh = w + hh * KSZ;
#pragma unroll
    for (int j = 0; j < KSZ; ++j) {
        int t = n + j - (KSZ / 2);
        if (t >= 0 && t < NSEQ) acc += (float)vb[(long long)t * DHD] * wh[j];
    }
    out[i] = acc;
}

// [B,H,N,64] f32 -> [B,N,512] f16
__global__ void merge_heads(const float* __restrict__ outhd, _Float16* __restrict__ ao)
{
    long long i = (long long)blockIdx.x * 256 + threadIdx.x;   // B*4096*512
    int c = (int)(i & 511);
    int n = (int)((i >> 9) & 4095);
    int b = (int)(i >> 21);
    int hh = c >> 6, d = c & 63;
    ao[i] = (_Float16)outhd[(((long long)b * NH + hh) * NSEQ + n) * DHD + d];
}

__global__ void residual_add(float* __restrict__ h, const float* __restrict__ proj,
                             const float* __restrict__ ob)
{
    long long i = (long long)blockIdx.x * 256 + threadIdx.x;
    h[i] += proj[i] + ob[i & 511];
}

// final layernorm on cls row + linear head + sigmoid + threshold
__global__ __launch_bounds__(256)
void classify(const float* __restrict__ hbuf, const float* __restrict__ sc,
              const float* __restrict__ bi, const float* __restrict__ fw,
              const float* __restrict__ fb, float* __restrict__ out)
{
    __shared__ float red[256];
    int b = blockIdx.x, t = threadIdx.x;
    const float* x = hbuf + (long long)b * NSEQ * DIMC;        // cls row (pos 0)
    float v0 = x[t], v1 = x[t + 256];
    red[t] = v0 + v1; __syncthreads();
    for (int s = 128; s > 0; s >>= 1) { if (t < s) red[t] += red[t + s]; __syncthreads(); }
    float mu = red[0] / (float)DIMC; __syncthreads();
    red[t] = (v0 - mu) * (v0 - mu) + (v1 - mu) * (v1 - mu); __syncthreads();
    for (int s = 128; s > 0; s >>= 1) { if (t < s) red[t] += red[t + s]; __syncthreads(); }
    float rs = rsqrtf(red[0] / (float)DIMC + 1e-5f); __syncthreads();
    float n0 = (v0 - mu) * rs * sc[t] + bi[t];
    float n1 = (v1 - mu) * rs * sc[t + 256] + bi[t + 256];
    red[t] = n0 * fw[t] + n1 * fw[t + 256]; __syncthreads();
    for (int s = 128; s > 0; s >>= 1) { if (t < s) red[t] += red[t + s]; __syncthreads(); }
    if (t == 0) {
        float logit = red[0] + fb[0];
        float prob = 1.0f / (1.0f + __expf(-logit));
        out[b] = logit;
        out[4 + b] = prob;
        ((int*)out)[8 + b] = prob > 0.5f ? 1 : 0;
    }
}

// ---------------------------------------------------------------------------
extern "C" void kernel_launch(void* const* d_in, const int* in_sizes, int n_in,
                              void* d_out, int out_size, void* d_ws, size_t ws_size,
                              hipStream_t stream)
{
    const float* data     = (const float*)d_in[0];
    const float* fc1_w    = (const float*)d_in[1];
    const float* fc1_b    = (const float*)d_in[2];
    const float* cls_tok  = (const float*)d_in[3];
    const float* ln_scale = (const float*)d_in[4];
    const float* ln_bias  = (const float*)d_in[5];
    const float* qkv_w    = (const float*)d_in[6];
    const float* out_w    = (const float*)d_in[7];
    const float* out_b    = (const float*)d_in[8];
    const float* conv_w   = (const float*)d_in[9];
    const float* fn_sc    = (const float*)d_in[10];
    const float* fn_bi    = (const float*)d_in[11];
    const float* fcout_w  = (const float*)d_in[12];
    const float* fcout_b  = (const float*)d_in[13];

    // ---- workspace carve-up (256B aligned) ----
    char* base = (char*)d_ws;
    size_t off = 0;
    auto alloc = [&](size_t bytes) -> char* {
        char* p = base + off;
        off += (bytes + 255) & ~(size_t)255;
        return p;
    };
    const size_t ROWS = (size_t)NB * NSEQ;                 // 16384
    _Float16* Xin   = (_Float16*)alloc(ROWS * INDIM * 2);
    _Float16* fc1wT = (_Float16*)alloc((size_t)DIMC * INDIM * 2);
    _Float16* qkvwT = (_Float16*)alloc((size_t)3 * DIMC * DIMC * 2);
    _Float16* outwT = (_Float16*)alloc((size_t)DIMC * DIMC * 2);
    float*    hbuf  = (float*)   alloc(ROWS * DIMC * 4);
    float*    Cbuf  = (float*)   alloc(ROWS * DIMC * 4);
    _Float16* lnbuf = (_Float16*)alloc(ROWS * DIMC * 2);
    _Float16* qkvb  = (_Float16*)alloc(ROWS * 3 * DIMC * 2);
    _Float16* qb    = (_Float16*)alloc(ROWS * DIMC * 2);
    _Float16* kb    = (_Float16*)alloc(ROWS * DIMC * 2);
    _Float16* vb    = (_Float16*)alloc(ROWS * DIMC * 2);
    _Float16* qlb   = (_Float16*)alloc((size_t)32 * NLM * DHD * 2);
    _Float16* klb   = (_Float16*)alloc((size_t)32 * NLM * DHD * 2);
    _Float16* big   = (_Float16*)alloc((size_t)32 * NSEQ * NLM * 2);  // attn3 then attn1
    _Float16* x2    = (_Float16*)alloc((size_t)32 * NLM * NLM * 2);
    _Float16* zA    = (_Float16*)alloc((size_t)32 * NLM * NLM * 2);
    _Float16* zB    = (_Float16*)alloc((size_t)32 * NLM * NLM * 2);
    _Float16* xzb   = (_Float16*)alloc((size_t)32 * NLM * NLM * 2);
    _Float16* tb0   = (_Float16*)alloc((size_t)32 * NLM * NLM * 2);
    _Float16* tb1   = (_Float16*)alloc((size_t)32 * NLM * NLM * 2);
    _Float16* Y1    = (_Float16*)alloc((size_t)32 * NLM * DHD * 2);
    _Float16* Y2    = (_Float16*)alloc((size_t)32 * NLM * DHD * 2);
    float*    outhd = (float*)   alloc((size_t)32 * NSEQ * DHD * 4);
    _Float16* aob   = (_Float16*)alloc(ROWS * DIMC * 2);
    float*    denom = (float*)   alloc(256);
    (void)ws_size; (void)in_sizes; (void)n_in; (void)out_size;

    // ---- stem: fc1 + relu + PE + cls ----
    pack_input<<<(int)(ROWS * INDIM / 256), 256, 0, stream>>>(data, Xin);
    convert_transpose<<<(INDIM * DIMC) / 256, 256, 0, stream>>>(fc1_w, fc1wT, INDIM, DIMC);
    wmma_gemm<true, false><<<dim3(DIMC / 64, (int)(ROWS / 128), 1), 256, 0, stream>>>(
        Xin, fc1wT, Cbuf, (int)ROWS, DIMC, INDIM, INDIM, INDIM, DIMC, 0, 0, 0, 1.0f);
    fc1_post<<<(int)(ROWS * DIMC / 256), 256, 0, stream>>>(Cbuf, fc1_b, cls_tok, hbuf);

    for (int L = 0; L < 2; ++L) {
        // layernorm -> f16
        layernorm_f16<<<(int)(ROWS / 8), 256, 0, stream>>>(
            hbuf, ln_scale + L * DIMC, ln_bias + L * DIMC, lnbuf);
        // qkv projection
        convert_transpose<<<(DIMC * 3 * DIMC) / 256, 256, 0, stream>>>(
            qkv_w + (size_t)L * DIMC * 3 * DIMC, qkvwT, DIMC, 3 * DIMC);
        wmma_gemm<true, true><<<dim3(3 * DIMC / 64, (int)(ROWS / 128), 1), 256, 0, stream>>>(
            lnbuf, qkvwT, qkvb, (int)ROWS, 3 * DIMC, DIMC, DIMC, DIMC, 3 * DIMC, 0, 0, 0, 1.0f);
        split_qkv<<<(int)(ROWS * DIMC / 256), 256, 0, stream>>>(qkvb, qb, kb, vb);
        landmarks<<<32 * NLM * DHD / 256, 256, 0, stream>>>(qb, kb, qlb, klb);

        // attn3 = softmax(q_l @ k^T)  [bh,256,4096]
        wmma_gemm<true, true><<<dim3(NSEQ / 64, 2, 32), 256, 0, stream>>>(
            qlb, kb, big, NLM, NSEQ, DHD, DHD, DHD, NSEQ,
            (long long)NLM * DHD, (long long)NSEQ * DHD, (long long)NLM * NSEQ, 1.0f);
        softmax_rows<<<32 * NLM, 256, 0, stream>>>(big, NSEQ);
        // Y1 = attn3 @ v  [bh,256,64]
        wmma_gemm<false, true><<<dim3(1, 2, 32), 256, 0, stream>>>(
            big, vb, Y1, NLM, DHD, NSEQ, NSEQ, DHD, DHD,
            (long long)NLM * NSEQ, (long long)NSEQ * DHD, (long long)NLM * DHD, 1.0f);

        // attn2 = softmax(q_l @ k_l^T)  [bh,256,256]
        wmma_gemm<true, true><<<dim3(NLM / 64, 2, 32), 256, 0, stream>>>(
            qlb, klb, x2, NLM, NLM, DHD, DHD, DHD, NLM,
            (long long)NLM * DHD, (long long)NLM * DHD, (long long)NLM * NLM, 1.0f);
        softmax_rows<<<32 * NLM, 256, 0, stream>>>(x2, NLM);

        // Newton-Schulz pinv (row sums == 1 post-softmax, so denom = max col-sum)
        init_denom<<<1, 1, 0, stream>>>(denom);
        colsum_max<<<32, 256, 0, stream>>>(x2, denom);
        pinv_init<<<32 * NLM * NLM / 256, 256, 0, stream>>>(x2, denom, zA);
        _Float16* zc = zA; _Float16* zn = zB;
        const long long SQ = (long long)NLM * NLM;
        for (int it = 0; it < 6; ++it) {
            wmma_gemm<false, true><<<dim3(NLM / 64, 2, 32), 256, 0, stream>>>(
                x2, zc, xzb, NLM, NLM, NLM, NLM, NLM, NLM, SQ, SQ, SQ, 1.0f);      // xz = x@z
            diag_sub<<<32 * NLM * NLM / 256, 256, 0, stream>>>(xzb, tb0, 7.0f);    // t1
            wmma_gemm<false, true><<<dim3(NLM / 64, 2, 32), 256, 0, stream>>>(
                xzb, tb0, tb1, NLM, NLM, NLM, NLM, NLM, NLM, SQ, SQ, SQ, 1.0f);    // xz@t1
            diag_sub<<<32 * NLM * NLM / 256, 256, 0, stream>>>(tb1, tb0, 15.0f);   // t2
            wmma_gemm<false, true><<<dim3(NLM / 64, 2, 32), 256, 0, stream>>>(
                xzb, tb0, tb1, NLM, NLM, NLM, NLM, NLM, NLM, SQ, SQ, SQ, 1.0f);    // xz@t2
            diag_sub<<<32 * NLM * NLM / 256, 256, 0, stream>>>(tb1, tb0, 13.0f);   // t3
            wmma_gemm<false, true><<<dim3(NLM / 64, 2, 32), 256, 0, stream>>>(
                zc, tb0, zn, NLM, NLM, NLM, NLM, NLM, NLM, SQ, SQ, SQ, 0.25f);     // z'
            _Float16* tmp = zc; zc = zn; zn = tmp;
        }

        // Y2 = pinv @ Y1
        wmma_gemm<false, true><<<dim3(1, 2, 32), 256, 0, stream>>>(
            zc, Y1, Y2, NLM, DHD, NLM, NLM, DHD, DHD,
            SQ, (long long)NLM * DHD, (long long)NLM * DHD, 1.0f);

        // attn1 = softmax(q @ k_l^T)  [bh,4096,256]  (reuse big buffer)
        wmma_gemm<true, true><<<dim3(NLM / 64, NSEQ / 128, 32), 256, 0, stream>>>(
            qb, klb, big, NSEQ, NLM, DHD, DHD, DHD, NLM,
            (long long)NSEQ * DHD, (long long)NLM * DHD, (long long)NSEQ * NLM, 1.0f);
        softmax_rows<<<32 * NSEQ, 256, 0, stream>>>(big, NLM);
        // out = attn1 @ Y2  [bh,4096,64] f32
        wmma_gemm<false, false><<<dim3(1, NSEQ / 128, 32), 256, 0, stream>>>(
            big, Y2, outhd, NSEQ, DHD, NLM, NLM, DHD, DHD,
            (long long)NSEQ * NLM, (long long)NLM * DHD, (long long)NSEQ * DHD, 1.0f);

        // + depthwise conv residual on v, merge heads, out projection, residual
        conv_add<<<32 * NSEQ * DHD / 256, 256, 0, stream>>>(
            vb, conv_w + (size_t)L * NH * KSZ, outhd);
        merge_heads<<<(int)(ROWS * DIMC / 256), 256, 0, stream>>>(outhd, aob);
        convert_transpose<<<(DIMC * DIMC) / 256, 256, 0, stream>>>(
            out_w + (size_t)L * DIMC * DIMC, outwT, DIMC, DIMC);
        wmma_gemm<true, false><<<dim3(DIMC / 64, (int)(ROWS / 128), 1), 256, 0, stream>>>(
            aob, outwT, Cbuf, (int)ROWS, DIMC, DIMC, DIMC, DIMC, DIMC, 0, 0, 0, 1.0f);
        residual_add<<<(int)(ROWS * DIMC / 256), 256, 0, stream>>>(
            hbuf, Cbuf, out_b + L * DIMC);
    }

    classify<<<NB, 256, 0, stream>>>(hbuf, fn_sc, fn_bi, fcout_w, fcout_b, (float*)d_out);
}
